// SetAttentionAggregator_12549894439643
// MI455X (gfx1250) — matrix operand order repaired and checked by reading, not compiled, not run on hardware
//
#include <hip/hip_runtime.h>
#include <hip/hip_bf16.h>
#include <math.h>

typedef __attribute__((ext_vector_type(8)))  float  v8f;
typedef __attribute__((ext_vector_type(16))) __bf16 v16bf;

#define LN_EPS 1e-5f

struct alignas(16) I4 { int x, y, z, w; };
union FragU { v16bf v; I4 i[2]; };

__device__ __forceinline__ float gelu_exact(float x) {
    return 0.5f * x * (1.0f + erff(x * 0.70710678118654752440f));
}
__device__ __forceinline__ v8f v8f_zero() {
    v8f z; for (int p = 0; p < 8; ++p) z[p] = 0.0f; return z;
}
// A fragment (16x32 bf16), source row-major, 32 bf16 per row. lane: m = lane&15, hi = lane>>4
// VGPR v(0..3): K = 2v + 8*hi ; VGPR v(4..7): K = 16 + 2(v-4) + 8*hi
__device__ __forceinline__ v16bf load_afrag(const __bf16* row, int hi) {
    FragU u;
    u.i[0] = *(const I4*)(row + 8 * hi);
    u.i[1] = *(const I4*)(row + 16 + 8 * hi);
    return u.v;
}
// B fragment (32x16 bf16), source column-major: per output col n, 32 bf16 contiguous.
// VGPR v(0..3): K = 2v + 8*hi ; VGPR v(4..7): K = 16 + 2(v-4) + 8*hi
__device__ __forceinline__ v16bf load_bfrag(const __bf16* col, int hi) {
    FragU u;
    u.i[0] = *(const I4*)(col + 8 * hi);
    u.i[1] = *(const I4*)(col + 16 + 8 * hi);
    return u.v;
}
__device__ __forceinline__ float bf16_lo(unsigned int u) { return __uint_as_float(u << 16); }
__device__ __forceinline__ float bf16_hi(unsigned int u) { return __uint_as_float(u & 0xffff0000u); }

// ---- CDNA5 async global->LDS copy: 64 contiguous bytes (4 x b128).
// INST_OFFSET is added to BOTH the LDS and global address, so one address pair
// covers the whole 64-B column. Tracked with ASYNCcnt.
__device__ __forceinline__ void async_copy64(unsigned lds_off, unsigned long long gaddr) {
    asm volatile(
        "global_load_async_to_lds_b128 %0, %1, off\n\t"
        "global_load_async_to_lds_b128 %0, %1, off offset:16\n\t"
        "global_load_async_to_lds_b128 %0, %1, off offset:32\n\t"
        "global_load_async_to_lds_b128 %0, %1, off offset:48"
        :: "v"(lds_off), "v"(gaddr) : "memory");
}
__device__ __forceinline__ void wait_asynccnt_le8() {
    asm volatile("s_wait_asynccnt 0x8" ::: "memory");
}
__device__ __forceinline__ void wait_asynccnt_0() {
    asm volatile("s_wait_asynccnt 0x0" ::: "memory");
}

// ---------------- k0: weight convert + transpose to bf16 ----------------
__global__ __launch_bounds__(256)
void k0_convert(const float* __restrict__ W1, const float* __restrict__ Wq1,
                const float* __restrict__ Wc,
                __bf16* __restrict__ W1t, __bf16* __restrict__ Wq1t,
                __bf16* __restrict__ Wct)
{
    int i = blockIdx.x * 256 + threadIdx.x;
    if (i < 512 * 512) {                       // W1 (k,n) -> W1t[n][k]
        int k = i >> 9, n = i & 511;
        W1t[n * 512 + k] = (__bf16)W1[i];
    } else if (i < 512 * 512 + 512 * 64) {     // Wq1 (k,h) -> Wq1t[h][k]
        int j = i - 512 * 512;
        int k = j >> 6, h = j & 63;
        Wq1t[h * 512 + k] = (__bf16)Wq1[j];
    } else if (i < 512 * 512 + 512 * 64 + 1536 * 512) {  // Wc (f,d) -> Wct[d][f]
        int j = i - (512 * 512 + 512 * 64);
        int f = j >> 9, d = j & 511;
        Wct[d * 1536 + f] = (__bf16)Wc[j];
    }
}

// ---------------- k1: x@W1 + b1 -> LN -> GELU -> mask -> t ; fused quality logits ----------------
__global__ __launch_bounds__(256)
void k1_gemm_ln_quality(const float* __restrict__ x,
                        const unsigned char* __restrict__ mask,
                        const __bf16* __restrict__ W1t,
                        const float* __restrict__ b1,
                        const float* __restrict__ g1,
                        const float* __restrict__ be1,
                        const __bf16* __restrict__ Wq1t,
                        const float* __restrict__ bq1,
                        const float* __restrict__ Wq2,
                        const float* __restrict__ bq2,
                        __bf16* __restrict__ t_out,
                        float* __restrict__ qlogit)
{
    extern __shared__ unsigned char smem[];
    // double-buffered tiles: Abuf = 2 x (32x32 bf16), Bbuf = 2 x (512x32 bf16, col-major)
    __bf16* AbufB = (__bf16*)(smem);              // buf b at AbufB + b*1024
    __bf16* BbufB = (__bf16*)(smem + 4096);       // buf b at BbufB + b*16384
    __bf16* tbf   = (__bf16*)(smem + 69632);      // 32 x 512 bf16 (32 KB)
    float*  qbuf  = (float*)(smem + 102400);      // 32 x 64 f32   (8 KB)
    float*  rS    = (float*)(smem + 110592);      // 32 x 4
    float*  rQ    = (float*)(smem + 111104);      // 32 x 4
    float*  musL  = (float*)(smem + 111616);      // 32
    float*  rstdL = (float*)(smem + 111744);      // 32
    float*  maskF = (float*)(smem + 111872);      // 32  (total 112000 B)

    const int tid  = threadIdx.x;
    const int wave = tid >> 5;
    const int lane = tid & 31;
    const int m16  = lane & 15;
    const int hi   = lane >> 4;
    const int wr   = wave >> 2;            // row tile 0..1 (16 rows each)
    const int wc   = wave & 3;             // col group 0..3 (128 cols each)
    const int colbase = wc * 128;
    const int row0 = blockIdx.x * 32;      // flat row (b*512+n)

    // per-thread staging coordinates
    const int ar = (tid * 4) >> 5, ak = (tid * 4) & 31;   // A: 4 consecutive f32
    const int c0 = tid * 2;                               // B: two 64-B columns

    auto stageA = [&](int kc, int buf) {
        const float4 v = *(const float4*)(x + (size_t)(row0 + ar) * 512 + kc * 32 + ak);
        __bf16* dst = AbufB + buf * 1024 + ar * 32 + ak;
        dst[0] = (__bf16)v.x; dst[1] = (__bf16)v.y;
        dst[2] = (__bf16)v.z; dst[3] = (__bf16)v.w;
    };
    auto stageB = [&](int kc, int buf) {   // 8 async b128 per thread -> ASYNCcnt += 8
        for (int c = c0; c < c0 + 2; ++c) {
            unsigned lds_off = (unsigned)(size_t)(BbufB + buf * 16384 + c * 32);
            unsigned long long g =
                (unsigned long long)(size_t)(W1t + (size_t)c * 512 + kc * 32);
            async_copy64(lds_off, g);
        }
    };

    v8f acc[8];
    for (int j = 0; j < 8; ++j) acc[j] = v8f_zero();

    // prologue: stage chunk 0
    stageA(0, 0);
    stageB(0, 0);

    for (int kc = 0; kc < 16; ++kc) {
        const int cur = kc & 1;
        __syncthreads();                       // all waves done reading buf[1-cur] (chunk kc-1)
        if (kc < 15) {                         // prefetch chunk kc+1 into the other buffer
            stageA(kc + 1, 1 - cur);
            stageB(kc + 1, 1 - cur);
            wait_asynccnt_le8();               // chunk kc complete; kc+1's 8 still in flight
        } else {
            wait_asynccnt_0();
        }
        __syncthreads();                       // publish chunk kc (async + ds stores)
        const __bf16* Ab = AbufB + cur * 1024;
        const __bf16* Bb = BbufB + cur * 16384;
        v16bf a = load_afrag(Ab + (16 * wr + m16) * 32, hi);
        for (int j = 0; j < 8; ++j) {
            v16bf b = load_bfrag(Bb + (colbase + 16 * j + m16) * 32, hi);
            acc[j] = __builtin_amdgcn_wmma_f32_16x16x32_bf16(
                false, a, false, b, (short)0, acc[j], false, false);
        }
    }

    // per-lane bias/affine params for its 8 columns
    float b1v[8], g1v[8], be1v[8];
    for (int j = 0; j < 8; ++j) {
        int n = colbase + 16 * j + m16;
        b1v[j] = b1[n]; g1v[j] = g1[n]; be1v[j] = be1[n];
    }
    // row statistics from accumulators (C layout: VGPR p -> row p+8*hi, lane -> col)
    for (int p = 0; p < 8; ++p) {
        float s = 0.f, q = 0.f;
        for (int j = 0; j < 8; ++j) {
            float v = acc[j][p] + b1v[j];
            s += v; q += v * v;
        }
        for (int m = 1; m < 16; m <<= 1) {
            s += __shfl_xor(s, m, 32);
            q += __shfl_xor(q, m, 32);
        }
        if (m16 == 0) {
            int lrow = 16 * wr + p + 8 * hi;
            rS[lrow * 4 + wc] = s;
            rQ[lrow * 4 + wc] = q;
        }
    }
    __syncthreads();
    if (tid < 32) {
        float s = rS[tid * 4] + rS[tid * 4 + 1] + rS[tid * 4 + 2] + rS[tid * 4 + 3];
        float q = rQ[tid * 4] + rQ[tid * 4 + 1] + rQ[tid * 4 + 2] + rQ[tid * 4 + 3];
        float mu = s * (1.0f / 512.0f);
        float var = q * (1.0f / 512.0f) - mu * mu;
        musL[tid]  = mu;
        rstdL[tid] = rsqrtf(var + LN_EPS);
        maskF[tid] = mask[row0 + tid] ? 1.0f : 0.0f;
    }
    __syncthreads();
    // LN + GELU + mask; write LDS bf16 copy and global t (bf16)
    for (int p = 0; p < 8; ++p) {
        int lrow = 16 * wr + p + 8 * hi;
        float mu = musL[lrow], rstd = rstdL[lrow], mf = maskF[lrow];
        for (int j = 0; j < 8; ++j) {
            int n = colbase + 16 * j + m16;
            float v = acc[j][p] + b1v[j];
            v = (v - mu) * rstd * g1v[j] + be1v[j];
            v = gelu_exact(v) * mf;
            __bf16 bv = (__bf16)v;
            tbf[lrow * 512 + n] = bv;
            t_out[(size_t)(row0 + lrow) * 512 + n] = bv;
        }
    }
    __syncthreads();

    // fused quality net GEMM: qh(32x64) = t(32x512) @ Wq1(512x64); one 16x16 tile per wave
    const int rt = wave & 1;
    const int ct = wave >> 1;
    v8f qacc = v8f_zero();
    for (int kc = 0; kc < 16; ++kc) {
        v16bf a = load_afrag(tbf + (16 * rt + m16) * 512 + kc * 32, hi);
        v16bf b = load_bfrag(Wq1t + (size_t)(16 * ct + m16) * 512 + kc * 32, hi);
        qacc = __builtin_amdgcn_wmma_f32_16x16x32_bf16(
            false, a, false, b, (short)0, qacc, false, false);
    }
    for (int p = 0; p < 8; ++p) {
        int lrow = 16 * rt + p + 8 * hi;
        qbuf[lrow * 64 + 16 * ct + m16] = qacc[p];
    }
    __syncthreads();
    if (tid < 32) {
        float s = 0.f;
        for (int h = 0; h < 64; ++h)
            s += gelu_exact(qbuf[tid * 64 + h] + bq1[h]) * Wq2[h];
        s += bq2[0];
        qlogit[row0 + tid] = mask[row0 + tid] ? s : -1e9f;
    }
}

// ---------------- k2: softmax over N + (max, mean, weighted) pools ----------------
__global__ __launch_bounds__(256)
void k2_pool(const __bf16* __restrict__ t,
             const float* __restrict__ qlogit,
             const unsigned char* __restrict__ mask,
             __bf16* __restrict__ combined)
{
    __shared__ float wgt[512];
    __shared__ float red[256];
    __shared__ float redc[256];
    const int b = blockIdx.x;
    const int tid = threadIdx.x;

    float l0 = qlogit[b * 512 + tid];
    float l1 = qlogit[b * 512 + tid + 256];
    red[tid] = fmaxf(l0, l1);
    __syncthreads();
    for (int s = 128; s > 0; s >>= 1) {
        if (tid < s) red[tid] = fmaxf(red[tid], red[tid + s]);
        __syncthreads();
    }
    float mx = red[0];
    __syncthreads();
    float e0 = __expf(l0 - mx), e1 = __expf(l1 - mx);
    wgt[tid] = e0; wgt[tid + 256] = e1;
    red[tid]  = e0 + e1;
    redc[tid] = (mask[b * 512 + tid] ? 1.f : 0.f) + (mask[b * 512 + tid + 256] ? 1.f : 0.f);
    __syncthreads();
    for (int s = 128; s > 0; s >>= 1) {
        if (tid < s) { red[tid] += red[tid + s]; redc[tid] += redc[tid + s]; }
        __syncthreads();
    }
    float inv_sum = 1.0f / red[0];
    float act = fmaxf(redc[0], 1.0f);
    __syncthreads();
    wgt[tid] *= inv_sum;
    wgt[tid + 256] *= inv_sum;
    __syncthreads();

    // each thread owns two adjacent d (one u32 = 2 bf16 per n)
    const unsigned int* tp = (const unsigned int*)(t + (size_t)b * 512 * 512);
    float mx0 = -3.0e38f, mx1 = -3.0e38f;
    float sm0 = 0.f, sm1 = 0.f, ws0 = 0.f, ws1 = 0.f;
    for (int n = 0; n < 512; ++n) {
        unsigned int u = tp[n * 256 + tid];
        float v0 = bf16_lo(u), v1 = bf16_hi(u);
        float w = wgt[n];
        mx0 = fmaxf(mx0, v0); mx1 = fmaxf(mx1, v1);
        sm0 += v0;            sm1 += v1;
        ws0 += w * v0;        ws1 += w * v1;
    }
    const int d0 = tid * 2;
    __bf16* cb = combined + (size_t)b * 1536;
    cb[d0]            = (__bf16)mx0;         cb[d0 + 1]        = (__bf16)mx1;
    cb[512 + d0]      = (__bf16)(sm0 / act); cb[512 + d0 + 1]  = (__bf16)(sm1 / act);
    cb[1024 + d0]     = (__bf16)ws0;         cb[1024 + d0 + 1] = (__bf16)ws1;
}

// ---------------- k3: combined @ Wc + bc -> LN -> GELU -> out ----------------
__global__ __launch_bounds__(256)
void k3_final(const __bf16* __restrict__ combined,
              const __bf16* __restrict__ Wct,
              const float* __restrict__ bc,
              const float* __restrict__ g2,
              const float* __restrict__ be2,
              float* __restrict__ out)
{
    extern __shared__ unsigned char smem[];
    __bf16* Abuf  = (__bf16*)smem;                  // 16 x 32 bf16 (1 KB)
    float*  fout  = (float*)(smem + 1024);          // 16 x 512 f32 (32 KB)
    float*  musL  = (float*)(smem + 33792);         // 16
    float*  rstdL = (float*)(smem + 33856);         // 16

    const int tid = threadIdx.x;
    const int wave = tid >> 5, lane = tid & 31;
    const int m16 = lane & 15, hi = lane >> 4;
    const int row0 = blockIdx.x * 16;

    v8f acc[4];
    for (int j = 0; j < 4; ++j) acc[j] = v8f_zero();

    for (int kc = 0; kc < 48; ++kc) {
        __syncthreads();
        {
            int e = tid * 2;
            int r = e >> 5, k = e & 31;
            const __bf16* src = combined + (size_t)(row0 + r) * 1536 + kc * 32 + k;
            Abuf[r * 32 + k]     = src[0];
            Abuf[r * 32 + k + 1] = src[1];
        }
        __syncthreads();
        v16bf a = load_afrag(Abuf + m16 * 32, hi);
        for (int j = 0; j < 4; ++j) {
            int n = wave * 64 + j * 16 + m16;
            v16bf b = load_bfrag(Wct + (size_t)n * 1536 + kc * 32, hi);
            acc[j] = __builtin_amdgcn_wmma_f32_16x16x32_bf16(
                false, a, false, b, (short)0, acc[j], false, false);
        }
    }
    for (int j = 0; j < 4; ++j) {
        int n = wave * 64 + j * 16 + m16;
        float bcv = bc[n];
        for (int p = 0; p < 8; ++p)
            fout[(p + 8 * hi) * 512 + n] = acc[j][p] + bcv;
    }
    __syncthreads();
    for (int rr = 0; rr < 2; ++rr) {    // wave w owns rows 2w, 2w+1
        int r = wave * 2 + rr;
        float s = 0.f, q = 0.f;
        for (int i = lane; i < 512; i += 32) {
            float v = fout[r * 512 + i];
            s += v; q += v * v;
        }
        for (int m = 1; m < 32; m <<= 1) {
            s += __shfl_xor(s, m, 32);
            q += __shfl_xor(q, m, 32);
        }
        if (lane == 0) {
            float mu = s * (1.0f / 512.0f);
            float var = q * (1.0f / 512.0f) - mu * mu;
            musL[r] = mu; rstdL[r] = rsqrtf(var + LN_EPS);
        }
    }
    __syncthreads();
    for (int i = 0; i < 32; ++i) {
        int e = tid + 256 * i;
        int r = e >> 9, n = e & 511;
        float v = fout[e];
        v = (v - musL[r]) * rstdL[r] * g2[n] + be2[n];
        out[(size_t)(row0 + r) * 512 + n] = gelu_exact(v);
    }
}

extern "C" void kernel_launch(void* const* d_in, const int* in_sizes, int n_in,
                              void* d_out, int out_size, void* d_ws, size_t ws_size,
                              hipStream_t stream)
{
    (void)in_sizes; (void)n_in; (void)out_size; (void)ws_size;
    const float* x            = (const float*)d_in[0];
    const unsigned char* mask = (const unsigned char*)d_in[1];
    const float* W1  = (const float*)d_in[2];
    const float* b1  = (const float*)d_in[3];
    const float* g1  = (const float*)d_in[4];
    const float* be1 = (const float*)d_in[5];
    const float* Wq1 = (const float*)d_in[6];
    const float* bq1 = (const float*)d_in[7];
    const float* Wq2 = (const float*)d_in[8];
    const float* bq2 = (const float*)d_in[9];
    const float* Wc  = (const float*)d_in[10];
    const float* bc  = (const float*)d_in[11];
    const float* g2  = (const float*)d_in[12];
    const float* be2 = (const float*)d_in[13];
    float* out = (float*)d_out;

    unsigned char* ws = (unsigned char*)d_ws;
    size_t off = 0;
    auto carve = [&](size_t bytes) -> void* {
        void* p = ws + off;
        off += (bytes + 255) & ~(size_t)255;
        return p;
    };
    __bf16* W1t  = (__bf16*)carve((size_t)512 * 512 * 2);
    __bf16* Wq1t = (__bf16*)carve((size_t)512 * 64 * 2);
    __bf16* Wct  = (__bf16*)carve((size_t)1536 * 512 * 2);
    __bf16* tbuf = (__bf16*)carve((size_t)65536 * 512 * 2);
    float*  qlog = (float*)carve((size_t)65536 * 4);
    __bf16* comb = (__bf16*)carve((size_t)128 * 1536 * 2);

    k0_convert<<<4224, 256, 0, stream>>>(W1, Wq1, Wc, W1t, Wq1t, Wct);
    k1_gemm_ln_quality<<<2048, 256, 112000, stream>>>(
        x, mask, W1t, b1, g1, be1, Wq1t, bq1, Wq2, bq2, tbuf, qlog);
    k2_pool<<<128, 256, 0, stream>>>(tbuf, qlog, mask, comb);
    k3_final<<<8, 256, 33920, stream>>>(comb, Wct, bc, g2, be2, out);
}
